// Classifier_64965675320014
// MI455X (gfx1250) — matrix operand order repaired and checked by reading, not compiled, not run on hardware
//
#include <hip/hip_runtime.h>

// ---------------------------------------------------------------------------
// Types for CDNA5 WMMA (wave32)
// ---------------------------------------------------------------------------
typedef __attribute__((ext_vector_type(16))) __bf16 v16bf;
typedef __attribute__((ext_vector_type(4)))  __bf16 v4bf;
typedef __attribute__((ext_vector_type(8)))  float  v8f;

#define TM 128
#define TN 64
#define TKK 32

// ---------------------------------------------------------------------------
// WMMA GEMM:  C[M,N] = concat(A|A2)[M,K] @ op(W) + bias
//   transW==0 : W is [N,K] row-major  (C = A @ W^T)
//   transW==1 : W is [K,N] row-major  (C = A @ W)
//   A columns [0,split) from A (lda), [split,K) from A2 (lda2; lda2==0
//   broadcasts a single row -- used for the t0 time-encoding row).
//   ep: 0 = store (+bias), 1 = store relu(+bias),
//       2 = fused sigmoid + squared-diff vs E + atomic reduction into lossOut
// REQUIRES: M % 128 == 0, N % 64 == 0, K % 32 == 0, split % 32 == 0
// Block: 256 threads = 8 waves; block tile 128x64; 4 WMMA tiles per wave.
// Double-buffered LDS staging (f32 -> bf16 on the fly).
// ---------------------------------------------------------------------------
__global__ __launch_bounds__(256)
void k_gemm(const float* __restrict__ A, int lda,
            const float* __restrict__ A2, int lda2, int split,
            const float* __restrict__ W, int ldw, int transW,
            const float* __restrict__ bias,
            float* __restrict__ C, int ldc,
            const float* __restrict__ E, int ldE,
            float* __restrict__ lossOut, float lossScale,
            int M, int N, int Ktot, int ep)
{
    __shared__ __bf16 As[2][TM * TKK];
    __shared__ __bf16 Bs[2][TN * TKK];
    __shared__ float  red[8];
    (void)M; (void)N;

    const int tid  = threadIdx.x;
    const int lane = tid & 31;
    const int wave = tid >> 5;
    const long m0  = (long)blockIdx.x * TM;
    const long n0  = (long)blockIdx.y * TN;
    const int wm   = (wave >> 1) * 32;          // 0,32,64,96
    const int wn   = (wave & 1) * 16;           // 0,16 (+32 for 2nd tile)
    const int fr   = lane & 15;                 // fragment row/col
    const int kbA  = (lane & 16) ? 8 : 0;       // A 16-bit split-K pattern
    const int kbB  = (lane & 16) ? 8 : 0;       // B 16-bit split-K pattern

    // staging geometry: float4 chunks over a 32-wide K panel
    const int srow = tid >> 3;                  // 0..31
    const int scol = (tid & 7) * 4;             // 0..28
    const int bkk  = tid >> 3;                  // transW: k within tile
    const int bnn  = (tid & 7) * 8;             // transW: n base

    v8f acc00 = {0,0,0,0,0,0,0,0}, acc01 = {0,0,0,0,0,0,0,0};
    v8f acc10 = {0,0,0,0,0,0,0,0}, acc11 = {0,0,0,0,0,0,0,0};

    float4 ra[4], rb[2];

    auto loadTile = [&](int k0) {
        const float* S; long sld; int sk;
        if (k0 < split) { S = A;  sld = lda;  sk = k0; }
        else            { S = A2; sld = lda2; sk = k0 - split; }
        #pragma unroll
        for (int i = 0; i < 4; ++i)
            ra[i] = *(const float4*)(S + (m0 + srow + i * 32) * sld + sk + scol);
        if (!transW) {
            rb[0] = *(const float4*)(W + (n0 + srow)      * (long)ldw + k0 + scol);
            rb[1] = *(const float4*)(W + (n0 + srow + 32) * (long)ldw + k0 + scol);
        } else {
            rb[0] = *(const float4*)(W + (long)(k0 + bkk) * ldw + n0 + bnn);
            rb[1] = *(const float4*)(W + (long)(k0 + bkk) * ldw + n0 + bnn + 4);
        }
    };

    auto cvt4 = [](const float4& v) {
        v4bf h;
        h[0] = (__bf16)v.x; h[1] = (__bf16)v.y;
        h[2] = (__bf16)v.z; h[3] = (__bf16)v.w;
        return h;
    };

    auto storeTile = [&](int buf) {
        #pragma unroll
        for (int i = 0; i < 4; ++i)
            *(v4bf*)&As[buf][(srow + i * 32) * TKK + scol] = cvt4(ra[i]);
        if (!transW) {
            *(v4bf*)&Bs[buf][srow * TKK + scol]        = cvt4(rb[0]);
            *(v4bf*)&Bs[buf][(srow + 32) * TKK + scol] = cvt4(rb[1]);
        } else {
            v4bf h0 = cvt4(rb[0]), h1 = cvt4(rb[1]);
            #pragma unroll
            for (int j = 0; j < 4; ++j) {
                Bs[buf][(bnn + j)     * TKK + bkk] = h0[j];
                Bs[buf][(bnn + 4 + j) * TKK + bkk] = h1[j];
            }
        }
    };

    const int nk = Ktot / TKK;
    loadTile(0);
    storeTile(0);
    __syncthreads();

    for (int kt = 0; kt < nk; ++kt) {
        const int  cur  = kt & 1;
        const bool more = (kt + 1) < nk;
        if (more) loadTile((kt + 1) * TKK);               // global loads for next tile
        if (kt + 2 < nk) {                                // prefetch K+2 A panel
            int pk = (kt + 2) * TKK;
            if (pk < split)
                __builtin_prefetch(A + (m0 + srow) * (long)lda + pk + scol, 0, 0);
        }

        // ---- fragments from LDS (ISA 7.12.2 layouts)
        const __bf16* a0p = &As[cur][(wm + fr)      * TKK + kbA];
        const __bf16* a1p = &As[cur][(wm + 16 + fr) * TKK + kbA];
        const __bf16* b0p = &Bs[cur][(wn + fr)      * TKK + kbB];
        const __bf16* b1p = &Bs[cur][(wn + 32 + fr) * TKK + kbB];
        v16bf af0, af1, bf0, bf1;
        #pragma unroll
        for (int i = 0; i < 8; ++i) {
            af0[i] = a0p[i]; af0[i + 8] = a0p[i + 16];
            af1[i] = a1p[i]; af1[i + 8] = a1p[i + 16];
        }
        #pragma unroll
        for (int i = 0; i < 8; ++i) { bf0[i] = b0p[i]; bf0[i + 8] = b0p[i + 16]; bf1[i] = b1p[i]; bf1[i + 8] = b1p[i + 16]; }

        acc00 = __builtin_amdgcn_wmma_f32_16x16x32_bf16(false, af0, false, bf0, (short)0, acc00, false, false);
        acc01 = __builtin_amdgcn_wmma_f32_16x16x32_bf16(false, af0, false, bf1, (short)0, acc01, false, false);
        acc10 = __builtin_amdgcn_wmma_f32_16x16x32_bf16(false, af1, false, bf0, (short)0, acc10, false, false);
        acc11 = __builtin_amdgcn_wmma_f32_16x16x32_bf16(false, af1, false, bf1, (short)0, acc11, false, false);

        if (more) storeTile(cur ^ 1);                     // fill the other buffer
        __syncthreads();
    }

    // ---- epilogue (C layout: VGPR r -> row r, +8 for upper half-wave)
    const long row0 = m0 + wm + ((lane & 16) ? 8 : 0);
    const long row1 = row0 + 16;
    const long col0 = n0 + wn + fr;
    const long col1 = col0 + 32;

    if (ep == 2) {
        float local = 0.f;
        #pragma unroll
        for (int r = 0; r < 8; ++r) {
            float s, d;
            s = 1.f / (1.f + __expf(-acc00[r])); d = E[(row0 + r) * ldE + col0] - s; local += d * d;
            s = 1.f / (1.f + __expf(-acc01[r])); d = E[(row0 + r) * ldE + col1] - s; local += d * d;
            s = 1.f / (1.f + __expf(-acc10[r])); d = E[(row1 + r) * ldE + col0] - s; local += d * d;
            s = 1.f / (1.f + __expf(-acc11[r])); d = E[(row1 + r) * ldE + col1] - s; local += d * d;
        }
        #pragma unroll
        for (int off = 16; off; off >>= 1) local += __shfl_xor(local, off, 32);
        if (lane == 0) red[wave] = local;
        __syncthreads();
        if (tid == 0) {
            float t = 0.f;
            #pragma unroll
            for (int w2 = 0; w2 < 8; ++w2) t += red[w2];
            atomicAdd(lossOut, t * lossScale);
        }
    } else {
        const float bv0 = bias ? bias[col0] : 0.f;
        const float bv1 = bias ? bias[col1] : 0.f;
        #pragma unroll
        for (int r = 0; r < 8; ++r) {
            float v00 = acc00[r] + bv0, v01 = acc01[r] + bv1;
            float v10 = acc10[r] + bv0, v11 = acc11[r] + bv1;
            if (ep == 1) {
                v00 = fmaxf(v00, 0.f); v01 = fmaxf(v01, 0.f);
                v10 = fmaxf(v10, 0.f); v11 = fmaxf(v11, 0.f);
            }
            C[(row0 + r) * ldc + col0] = v00;
            C[(row0 + r) * ldc + col1] = v01;
            C[(row1 + r) * ldc + col0] = v10;
            C[(row1 + r) * ldc + col1] = v11;
        }
    }
}

// ---------------------------------------------------------------------------
// Time encoding: out[r,k,h] = cos((tsrc[r]-tngh[r*K+k])*freq[h] + phase[h])
// ---------------------------------------------------------------------------
__global__ void k_te(const float* __restrict__ tsrc, const float* __restrict__ tngh,
                     const float* __restrict__ freq, const float* __restrict__ phase,
                     float* __restrict__ out, long total)
{
    long i = (long)blockIdx.x * blockDim.x + threadIdx.x;
    if (i >= total) return;
    int  hh = (int)(i & 127);
    long rk = i >> 7;
    long r  = rk / 20;
    float dt = tsrc[r] - tngh[rk];
    out[i] = __cosf(dt * freq[hh] + phase[hh]);
}

__global__ void k_t0(const float* __restrict__ phase, float* __restrict__ t0row)
{
    int h = threadIdx.x;
    if (h < 128) t0row[h] = __cosf(phase[h]);
}

// ---------------------------------------------------------------------------
// Fused attention core: one wave per (node, head). K=20, dk=128, 2 heads.
// ---------------------------------------------------------------------------
__global__ __launch_bounds__(256)
void k_attn(const float* __restrict__ qh, const float* __restrict__ kh,
            const float* __restrict__ vh, const int* __restrict__ idx,
            float* __restrict__ out, int N)
{
    const int lane = threadIdx.x & 31;
    const int wave = threadIdx.x >> 5;
    const int g = blockIdx.x * 8 + wave;
    const int n = g >> 1;
    const int h = g & 1;
    if (n >= N) return;
    const int dbase = h * 128 + lane * 4;

    const float4 q = *(const float4*)(qh + (long)n * 256 + dbase);
    const float scale = 0.08838834764831845f;   // 1/sqrt(128)
    float s[20];

    #pragma unroll
    for (int k = 0; k < 20; ++k) {
        const float4 kv = *(const float4*)(kh + ((long)n * 20 + k) * 256 + dbase);
        float p = q.x * kv.x + q.y * kv.y + q.z * kv.z + q.w * kv.w;
        #pragma unroll
        for (int off = 16; off; off >>= 1) p += __shfl_xor(p, off, 32);
        s[k] = p * scale;
    }
    #pragma unroll
    for (int k = 0; k < 20; ++k)
        if (idx[(long)n * 20 + k] == 0) s[k] = -1e10f;

    float mx = s[0];
    #pragma unroll
    for (int k = 1; k < 20; ++k) mx = fmaxf(mx, s[k]);
    float sum = 0.f;
    #pragma unroll
    for (int k = 0; k < 20; ++k) { s[k] = __expf(s[k] - mx); sum += s[k]; }
    const float inv = 1.f / sum;

    float4 acc = {0.f, 0.f, 0.f, 0.f};
    #pragma unroll
    for (int k = 0; k < 20; ++k) {
        float a = s[k] * inv;
        const float4 vv = *(const float4*)(vh + ((long)n * 20 + k) * 256 + dbase);
        acc.x += a * vv.x; acc.y += a * vv.y; acc.z += a * vv.z; acc.w += a * vv.w;
    }
    *(float4*)(out + (long)n * 256 + dbase) = acc;
}

// ---------------------------------------------------------------------------
// Residual + LayerNorm over D=256: x = fcout + concat(src_lin, t0row); LN(x)
// ---------------------------------------------------------------------------
__global__ __launch_bounds__(256)
void k_ln(const float* __restrict__ fco, const float* __restrict__ srcl,
          const float* __restrict__ t0row, const float* __restrict__ g,
          const float* __restrict__ b, float* __restrict__ out, int N)
{
    const int lane = threadIdx.x & 31;
    const int wave = threadIdx.x >> 5;
    const int n = blockIdx.x * 8 + wave;
    if (n >= N) return;

    float x[8];
    #pragma unroll
    for (int i = 0; i < 8; ++i) {
        int d = lane * 8 + i;
        float qv = (d < 128) ? srcl[(long)n * 128 + d] : t0row[d - 128];
        x[i] = fco[(long)n * 256 + d] + qv;
    }
    float m = 0.f;
    #pragma unroll
    for (int i = 0; i < 8; ++i) m += x[i];
    #pragma unroll
    for (int off = 16; off; off >>= 1) m += __shfl_xor(m, off, 32);
    m *= (1.f / 256.f);
    float v = 0.f;
    #pragma unroll
    for (int i = 0; i < 8; ++i) { float d = x[i] - m; v += d * d; }
    #pragma unroll
    for (int off = 16; off; off >>= 1) v += __shfl_xor(v, off, 32);
    v *= (1.f / 256.f);
    const float inv = rsqrtf(v + 1e-5f);
    #pragma unroll
    for (int i = 0; i < 8; ++i) {
        int d = lane * 8 + i;
        out[(long)n * 256 + d] = (x[i] - m) * inv * g[d] + b[d];
    }
}

// ---------------------------------------------------------------------------
// Host-side helpers
// ---------------------------------------------------------------------------
static void gemm(hipStream_t s,
                 const float* A, int lda, const float* A2, int lda2, int split,
                 const float* W, int ldw, int transW, const float* bias,
                 float* C, int ldc, int M, int N, int K, int ep = 0,
                 const float* E = nullptr, int ldE = 0,
                 float* loss = nullptr, float lscale = 0.f)
{
    dim3 grid((M + TM - 1) / TM, (N + TN - 1) / TN);
    k_gemm<<<grid, 256, 0, s>>>(A, lda, A2, lda2, split, W, ldw, transW, bias,
                                C, ldc, E, ldE, loss, lscale, M, N, K, ep);
}

struct AttnW {
    const float *wq, *wk, *wv, *fcw, *fcb, *lng, *lnb, *m1w, *m1b, *m2w, *m2b;
};

static void run_attn(hipStream_t s, int N,
                     const float* srcl, const float* t0row,
                     const float* nghl, const float* t, const int* idx,
                     const AttnW& w, float* out,
                     float* qh, float* kh, float* vh,
                     float* ao, float* fco, float* lno, float* h1)
{
    // q/k/v projections (concat handled inside the GEMM)
    gemm(s, srcl, 128, t0row, 0,   128, w.wq, 256, 0, nullptr, qh, 256, N,       256, 256);
    gemm(s, nghl, 128, t,     128, 128, w.wk, 256, 0, nullptr, kh, 256, N * 20,  256, 256);
    gemm(s, nghl, 128, t,     128, 128, w.wv, 256, 0, nullptr, vh, 256, N * 20,  256, 256);
    // fused scores/softmax/weighted-sum
    k_attn<<<dim3((N * 2 + 7) / 8), 256, 0, s>>>(qh, kh, vh, idx, ao, N);
    // output fc + residual + layernorm
    gemm(s, ao, 256, nullptr, 0, 256, w.fcw, 256, 0, w.fcb, fco, 256, N, 256, 256);
    k_ln<<<dim3((N + 7) / 8), 256, 0, s>>>(fco, srcl, t0row, w.lng, w.lnb, lno, N);
    // merge MLP: relu(fc1(cat[ln, src])) -> fc2
    gemm(s, lno, 256, srcl, 128, 256, w.m1w, 384, 0, w.m1b, h1,  128, N, 128, 384, 1);
    gemm(s, h1,  128, nullptr, 0, 128, w.m2w, 128, 0, w.m2b, out, 128, N, 128, 128);
}

extern "C" void kernel_launch(void* const* d_in, const int* in_sizes, int n_in,
                              void* d_out, int out_size, void* d_ws, size_t ws_size,
                              hipStream_t stream)
{
    constexpr int B = 2048, K = 20, H = 128, D = 256, BK = B * K;
    constexpr int NAE = 4096, NCH = 2048, NCHUNK = BK / NCH;   // 20 chunks

    const float* src_feat  = (const float*)d_in[0];
    const float* src_t     = (const float*)d_in[1];
    const int*   ngh1_idx  = (const int*)  d_in[2];
    const float* ngh1_t    = (const float*)d_in[3];
    const float* ngh1_feat = (const float*)d_in[4];
    const int*   ngh2_idx  = (const int*)  d_in[5];
    const float* ngh2_t    = (const float*)d_in[6];
    const float* ngh2_feat = (const float*)d_in[7];
    const float* ae_x      = (const float*)d_in[8];
    const float* ae_adj    = (const float*)d_in[9];
    const float* W_lin     = (const float*)d_in[10];
    const float* b_lin     = (const float*)d_in[11];
    const float* freq      = (const float*)d_in[12];
    const float* phase     = (const float*)d_in[13];
    const float* awq       = (const float*)d_in[14];
    const float* awk       = (const float*)d_in[15];
    const float* awv       = (const float*)d_in[16];
    const float* afcw      = (const float*)d_in[17];
    const float* afcb      = (const float*)d_in[18];
    const float* alng      = (const float*)d_in[19];
    const float* alnb      = (const float*)d_in[20];
    const float* m1w       = (const float*)d_in[21];
    const float* m1b       = (const float*)d_in[22];
    const float* m2w       = (const float*)d_in[23];
    const float* m2b       = (const float*)d_in[24];
    const float* W_gc      = (const float*)d_in[25];
    const float* b_gc      = (const float*)d_in[26];
    (void)in_sizes; (void)n_in; (void)out_size; (void)ws_size;

    float* out   = (float*)d_out;            // src_embed [B,128]
    float* lossp = out + (long)B * H;        // recon_loss scalar

    // ---- workspace layout (floats)
    float* ws       = (float*)d_ws;
    float* t0row    = ws;                         // 128
    float* src_lin  = t0row   + 128;              // B*128
    float* ngh1_lin = src_lin + (long)B * H;      // BK*128
    float* t1       = ngh1_lin + (long)BK * H;    // BK*128
    float* src_l1   = t1       + (long)BK * H;    // B*128
    float* ngh_l1   = src_l1   + (long)B * H;     // BK*128
    float* gcxw     = ngh_l1   + (long)BK * H;    // NAE*128
    float* gcz      = gcxw     + (long)NAE * H;   // NAE*128
    // chunk-reused scratch
    float* feat2    = gcz      + (long)NAE * H;   // NCH*K*128
    float* t2c      = feat2    + (long)NCH * K * H;
    float* qh       = t2c      + (long)NCH * K * H;   // NCH*256
    float* kh       = qh       + (long)NCH * D;       // NCH*K*256
    float* vh       = kh       + (long)NCH * K * D;
    float* ao       = vh       + (long)NCH * K * D;   // NCH*256
    float* fco      = ao       + (long)NCH * D;
    float* lno      = fco      + (long)NCH * D;
    float* h1       = lno      + (long)NCH * D;       // NCH*128

    AttnW L0 = { awq, awk, awv, afcw, afcb, alng, alnb, m1w, m1b, m2w, m2b };
    AttnW L1 = { awq + (long)D * D, awk + (long)D * D, awv + (long)D * D,
                 afcw + (long)D * D, afcb + D, alng + D, alnb + D,
                 m1w + (long)H * (D + H), m1b + H, m2w + (long)H * H, m2b + H };

    // ---- shared precompute
    k_t0<<<1, 128, 0, stream>>>(phase, t0row);
    {
        long tot = (long)B * K * H;
        k_te<<<dim3((unsigned)((tot + 255) / 256)), 256, 0, stream>>>(
            src_t, ngh1_t, freq, phase, t1, tot);
    }
    gemm(stream, src_feat,  128, nullptr, 0, 128, W_lin, 128, 0, b_lin, src_lin,  128, B,  128, 128);
    gemm(stream, ngh1_feat, 128, nullptr, 0, 128, W_lin, 128, 0, b_lin, ngh1_lin, 128, BK, 128, 128);

    // ---- layer-1 embedding of src: attn layer 0
    run_attn(stream, B, src_lin, t0row, ngh1_lin, t1, ngh1_idx, L0, src_l1,
             qh, kh, vh, ao, fco, lno, h1);

    // ---- layer-1 embedding of each 1-hop neighbor (chunked over BK rows)
    for (int c = 0; c < NCHUNK; ++c) {
        const long base = (long)c * NCH;   // row offset in BK
        gemm(stream, ngh2_feat + base * K * H, 128, nullptr, 0, 128,
             W_lin, 128, 0, b_lin, feat2, 128, NCH * K, 128, 128);
        long tot = (long)NCH * K * H;
        k_te<<<dim3((unsigned)((tot + 255) / 256)), 256, 0, stream>>>(
            ngh1_t + base, ngh2_t + base * K, freq, phase, t2c, tot);
        run_attn(stream, NCH, ngh1_lin + base * H, t0row, feat2, t2c,
                 ngh2_idx + base * K, L0, ngh_l1 + base * H,
                 qh, kh, vh, ao, fco, lno, h1);
    }

    // ---- top-level attention (layer 1) -> src_embed in d_out
    run_attn(stream, B, src_l1, t0row, ngh_l1, t1, ngh1_idx, L1, out,
             qh, kh, vh, ao, fco, lno, h1);

    // ---- autoencoder: gc_z = adj @ (x @ W_gc) + b ; loss = mean((adj - sigmoid(z z^T))^2)
    gemm(stream, ae_x,   128,  nullptr, 0, 128,  W_gc, 128, 1, nullptr, gcxw, 128, NAE, 128, 128);
    gemm(stream, ae_adj, NAE,  nullptr, 0, NAE,  gcxw, 128, 1, b_gc,    gcz,  128, NAE, 128, NAE);
    hipMemsetAsync(lossp, 0, sizeof(float), stream);
    gemm(stream, gcz, 128, nullptr, 0, 128, gcz, 128, 0, nullptr,
         nullptr, 0, NAE, NAE, 128, /*ep=*/2,
         ae_adj, NAE, lossp, 1.f / ((float)NAE * (float)NAE));
}
